// DecoderGRU_42880953484298
// MI455X (gfx1250) — compile-verified, matrix operand order checked by reading, not run on hardware
//
#include <hip/hip_runtime.h>

#define TT 120
#define BB 32
#define EE 300
#define HH 256
#define VV 32000
#define CC 512
#define KX 576   // E+H = 556 padded to 18*32

#if defined(__has_builtin)
#  if __has_builtin(__builtin_amdgcn_tensor_load_to_lds) && __has_builtin(__builtin_amdgcn_s_wait_tensorcnt)
#    define HAVE_TDM 1
#  endif
#endif
#ifndef HAVE_TDM
#  define HAVE_TDM 0
#endif

typedef __attribute__((ext_vector_type(16))) __bf16 v16bf;
typedef __attribute__((ext_vector_type(8)))  float  v8f;
typedef unsigned int u32x4 __attribute__((ext_vector_type(4)));
typedef int          i32x8 __attribute__((ext_vector_type(8)));
typedef int          i32x4 __attribute__((ext_vector_type(4)));

__device__ __forceinline__ unsigned short f2bf(float x) {
  unsigned int u = __float_as_uint(x);
  u += 0x7FFFu + ((u >> 16) & 1u);   // round-to-nearest-even
  return (unsigned short)(u >> 16);
}

union FragBF { v16bf v; float4 q[2]; };

// CDNA5 16-bit 16x32 A/B fragment: lane L<16 holds row (L) K {k0..k0+7, k0+16..k0+23},
// lane L>=16 holds row (L-16) K {k0+8..k0+15, k0+24..k0+31}. Row-major [16][ld] source,
// K contiguous -> two 16-byte loads per lane.
__device__ __forceinline__ v16bf load_frag(const unsigned short* base, int ld,
                                           int row, int k0, int hi8) {
  FragBF u;
  const unsigned short* p = base + (size_t)row * ld + k0 + hi8;
  u.q[0] = *reinterpret_cast<const float4*>(p);
  u.q[1] = *reinterpret_cast<const float4*>(p + 16);
  return u.v;
}

#if HAVE_TDM
// 1D TDM transfer: 32 KB contiguous (4096 x 8-byte units) global -> LDS.
// This toolchain's builtin takes 6 args: (g0, g1, g2, g3, g4, cpol).
__device__ __forceinline__ void tdm_issue_32k(const void* gsrc, unsigned lds_byte_addr) {
  unsigned long long ga = (unsigned long long)(uintptr_t)gsrc;
  u32x4 g0;
  g0[0] = 1u;                                  // count=1, user descriptor
  g0[1] = lds_byte_addr;                       // lds_addr (bytes)
  g0[2] = (unsigned)ga;                        // global_addr[31:0]
  g0[3] = (unsigned)((ga >> 32) & 0x1FFFFFFu)  // global_addr[56:32]
          | 0x80000000u;                       // type=2 ("image")
  i32x8 g1;
  g1[0] = 0x00030000;                          // data_size=3 (8B units)
  g1[1] = (int)(4096u << 16);                  // tensor_dim0 = 4096 (bits 79:48)
  g1[2] = (int)(1u << 16);                     // tensor_dim1 = 1 (bits 111:80)
  g1[3] = (int)(4096u << 16);                  // tile_dim0 = 4096 (bits 127:112)
  g1[4] = 0;                                   // tile_dim1 = 0 (1D), tile_dim2 = 0
  g1[5] = 4096;                                // tensor_dim0_stride
  g1[6] = 0;
  g1[7] = 0;
  i32x4 z4 = {0, 0, 0, 0};
  i32x8 z8 = {0, 0, 0, 0, 0, 0, 0, 0};
  __builtin_amdgcn_tensor_load_to_lds(g0, g1, z4, z4, z8, 0);
}
#endif

__global__ void __launch_bounds__(256) k_f32_to_bf16(const float* __restrict__ s,
                                                     unsigned short* __restrict__ d, int n) {
  for (int i = blockIdx.x * blockDim.x + threadIdx.x; i < n; i += gridDim.x * blockDim.x)
    d[i] = f2bf(s[i]);
}

__global__ void __launch_bounds__(256) k_pad_wih(const float* __restrict__ s,
                                                 unsigned short* __restrict__ d) {
  int n = 768 * KX;
  for (int i = blockIdx.x * blockDim.x + threadIdx.x; i < n; i += gridDim.x * blockDim.x) {
    int r = i / KX, k = i - r * KX;
    d[i] = (k < EE + HH) ? f2bf(s[(size_t)r * (EE + HH) + k]) : (unsigned short)0;
  }
}

// mean over 49 spatial positions: features [B][C][49]
__global__ void __launch_bounds__(256) k_meanf(const float* __restrict__ feats,
                                               float* __restrict__ meanf) {
  int i = blockIdx.x * blockDim.x + threadIdx.x;  // b*512 + c
  if (i >= BB * CC) return;
  const float* p = feats + (size_t)i * 49;
  float s = 0.f;
  #pragma unroll
  for (int g = 0; g < 49; ++g) s += p[g];
  meanf[i] = s * (1.f / 49.f);
}

__global__ void __launch_bounds__(256) k_h0(const float* __restrict__ meanf,
                                            const float* __restrict__ wi,
                                            const float* __restrict__ bi,
                                            float* __restrict__ h0) {
  int i = blockIdx.x * blockDim.x + threadIdx.x;  // b*256 + h
  if (i >= BB * HH) return;
  int b = i >> 8, h = i & 255;
  const float* m = meanf + b * CC;
  const float* w = wi + (size_t)h * CC;
  float s = 0.f;
  for (int c = 0; c < CC; ++c) s += m[c] * w[c];
  h0[i] = s + bi[h];
}

// feat[b][h] = sum_{g,c} features[b][c][g] * W_fc2[h][g*512+c]  (wave per output)
__global__ void __launch_bounds__(256) k_feat(const float* __restrict__ feats,
                                              const float* __restrict__ w2,
                                              const float* __restrict__ b2,
                                              float* __restrict__ feat) {
  int wid = (blockIdx.x * blockDim.x + threadIdx.x) >> 5;
  int lane = threadIdx.x & 31;
  if (wid >= BB * HH) return;
  int b = wid >> 8, h = wid & 255;
  const float* fb = feats + (size_t)b * (CC * 49);
  const float* wr = w2 + (size_t)h * (CC * 49);
  float s = 0.f;
  for (int j = lane; j < CC * 49; j += 32) {
    int g = j >> 9, c = j & 511;           // flat = g*512 + c
    s += fb[c * 49 + g] * wr[j];
  }
  for (int o = 16; o > 0; o >>= 1) s += __shfl_down(s, o, 32);
  if (lane == 0) feat[wid] = s + b2[h];
}

// X[t*B+b][k] = concat(emb[b][t][:], feat[b][:], zero-pad) in bf16
__global__ void __launch_bounds__(256) k_build_x(const float* __restrict__ emb,
                                                 const float* __restrict__ feat,
                                                 unsigned short* __restrict__ x) {
  int n = TT * BB * KX;
  for (int i = blockIdx.x * blockDim.x + threadIdx.x; i < n; i += gridDim.x * blockDim.x) {
    int row = i / KX, k = i - row * KX;
    int t = row >> 5, b = row & 31;
    float v;
    if (k < EE)            v = emb[((size_t)b * TT + t) * EE + k];
    else if (k < EE + HH)  v = feat[b * HH + (k - EE)];
    else                   v = 0.f;
    x[i] = f2bf(v);
  }
}

// gi[T*B][768] = X @ W_ih^T + b_ih   (wave per 16x16 tile, K=576)
__global__ void __launch_bounds__(256) k_gi(const unsigned short* __restrict__ x,
                                            const unsigned short* __restrict__ wih,
                                            const float* __restrict__ bih,
                                            float* __restrict__ gi) {
  int tile = blockIdx.x * 8 + (threadIdx.x >> 5);       // 240*48 tiles, exact
  int lane = threadIdx.x & 31;
  int r = lane & 15;
  int hi8 = (lane >> 4) << 3;
  int mt = tile / 48, nt = tile - mt * 48;
  int m0 = mt * 16, n0 = nt * 16;
  v8f c = {};
  for (int k0 = 0; k0 < KX; k0 += 32) {
    v16bf a = load_frag(x,   KX, m0 + r, k0, hi8);
    v16bf b = load_frag(wih, KX, n0 + r, k0, hi8);
    c = __builtin_amdgcn_wmma_f32_16x16x32_bf16(false, a, false, b, (short)0, c, false, false);
  }
  float bias = bih[n0 + r];
  #pragma unroll
  for (int e = 0; e < 8; ++e) {
    int row = m0 + e + hi8;                              // C/D: lanes>=16 -> M+8
    gi[(size_t)row * 768 + n0 + r] = c[e] + bias;
  }
}

// Persistent sequential GRU scan: one workgroup (32 waves), h lives in LDS.
__global__ void __launch_bounds__(1024) k_gru(const float* __restrict__ h0,
                                              const float* __restrict__ gi,
                                              const unsigned short* __restrict__ whh,
                                              const float* __restrict__ bhh,
                                              unsigned short* __restrict__ hsb) {
  __shared__ float h32[BB * HH];            // 32 KB
  __shared__ unsigned short hb[BB * HH];    // 16 KB
  __shared__ float gh[BB * 768];            // 96 KB
  int tid = threadIdx.x;
  for (int i = tid; i < BB * HH; i += 1024) {
    float v = h0[i];
    h32[i] = v; hb[i] = f2bf(v);
  }
  __syncthreads();
  int wave = tid >> 5, lane = tid & 31;
  int r = lane & 15, hi8 = (lane >> 4) << 3;
  int tile0 = wave * 3;                      // 48%3==0 -> all 3 tiles share m-tile
  int mt = tile0 / 48;
  int m0 = mt * 16;
  for (int t = 0; t < TT; ++t) {
    v16bf a[8];
    #pragma unroll
    for (int kk = 0; kk < 8; ++kk)
      a[kk] = load_frag(hb, HH, m0 + r, kk * 32, hi8);
    #pragma unroll
    for (int i = 0; i < 3; ++i) {
      int n0 = ((tile0 + i) - mt * 48) * 16;
      v8f c = {};
      #pragma unroll
      for (int kk = 0; kk < 8; ++kk) {
        v16bf b = load_frag(whh, HH, n0 + r, kk * 32, hi8);
        c = __builtin_amdgcn_wmma_f32_16x16x32_bf16(false, a[kk], false, b, (short)0, c, false, false);
      }
      #pragma unroll
      for (int e = 0; e < 8; ++e)
        gh[(m0 + e + hi8) * 768 + n0 + r] = c[e];
    }
    __syncthreads();
    const float* git = gi + (size_t)t * (BB * 768);
    for (int i = tid; i < BB * HH; i += 1024) {
      int b = i >> 8, j = i & 255;
      const float* g   = git + b * 768;
      const float* ghb = gh + b * 768;
      float rg = 1.f / (1.f + __expf(-(g[j]       + ghb[j]       + bhh[j])));
      float zg = 1.f / (1.f + __expf(-(g[256 + j] + ghb[256 + j] + bhh[256 + j])));
      float ng = tanhf(g[512 + j] + rg * (ghb[512 + j] + bhh[512 + j]));
      float hn = (1.f - zg) * ng + zg * h32[i];
      h32[i] = hn;
      unsigned short hv = f2bf(hn);
      hb[i] = hv;
      hsb[(size_t)t * (BB * HH) + i] = hv;
    }
    __syncthreads();
  }
}

// Vocab GEMM: logits[b][t][v] = hs[t*B+b] @ W_fc^T + b_fc.
// Block = 8 waves = 128M x 64N macro-tile; B panel (32 KB) staged in LDS and
// shared by all 8 waves (8x less B traffic). A (full K=256) lives in registers.
// B panels are DMA'd by the Tensor Data Mover, double-buffered across the
// block's 5 n-groups; s_wait_tensorcnt + workgroup barrier synchronize.
#define NGRP 5   // n-groups (64 cols each) per block: 100 blocks * 5 * 64 = 32000
__global__ void __launch_bounds__(256) k_logits(const unsigned short* __restrict__ hsb,
                                                const unsigned short* __restrict__ wfc,
                                                const float* __restrict__ bfc,
                                                float* __restrict__ out) {
  __shared__ __align__(16) unsigned short bbuf[2][64 * HH];   // 2 x 32 KB
  int tid = threadIdx.x;
  int wave = tid >> 5, lane = tid & 31;
  int r = lane & 15, hi8 = (lane >> 4) << 3;

  int mgrp = blockIdx.y;                  // 0..29
  int nblk = blockIdx.x;                  // 0..99
  int m0 = (mgrp * 8 + wave) * 16;
  int t = m0 >> 5;                        // constant within a 16-row tile
  int g0row = nblk * (NGRP * 64);

  // A fragments (full K) for this wave's m-tile stay in registers.
  v16bf a[8];
  #pragma unroll
  for (int kk = 0; kk < 8; ++kk)
    a[kk] = load_frag(hsb, HH, m0 + r, kk * 32, hi8);

#if HAVE_TDM
  if (wave == 0)
    tdm_issue_32k(wfc + (size_t)g0row * HH, (unsigned)(uintptr_t)(const void*)bbuf[0]);
#endif

  for (int g = 0; g < NGRP; ++g) {
    int cur = g & 1;
#if HAVE_TDM
    if (wave == 0) __builtin_amdgcn_s_wait_tensorcnt(0);
    __syncthreads();                                   // panel g ready for all waves
    if (wave == 0 && g + 1 < NGRP)                     // prefetch next panel (overlaps WMMA)
      tdm_issue_32k(wfc + (size_t)(g0row + (g + 1) * 64) * HH,
                    (unsigned)(uintptr_t)(const void*)bbuf[cur ^ 1]);
#else
    cur = 0;
    __syncthreads();                                   // previous consumers done
    {
      const float4* s = (const float4*)(wfc + (size_t)(g0row + g * 64) * HH);
      float4* d = (float4*)bbuf[0];
      for (int i = tid; i < 2048; i += 256) d[i] = s[i];
    }
    __syncthreads();
#endif
    const unsigned short* bl = bbuf[cur];
    int nbase = g0row + g * 64;
    for (int i = 0; i < 4; ++i) {
      v8f c = {};
      #pragma unroll
      for (int kk = 0; kk < 8; ++kk) {
        v16bf b = load_frag(bl, HH, i * 16 + r, kk * 32, hi8);
        c = __builtin_amdgcn_wmma_f32_16x16x32_bf16(false, a[kk], false, b, (short)0, c, false, false);
      }
      int v = nbase + i * 16 + r;
      float bias = bfc[v];
      #pragma unroll
      for (int e = 0; e < 8; ++e) {
        int row = m0 + e + hi8;
        int b = row & 31;
        __builtin_nontemporal_store(c[e] + bias, &out[((size_t)b * TT + t) * VV + v]);
      }
    }
  }
}

extern "C" void kernel_launch(void* const* d_in, const int* in_sizes, int n_in,
                              void* d_out, int out_size, void* d_ws, size_t ws_size,
                              hipStream_t stream) {
  (void)in_sizes; (void)n_in; (void)out_size; (void)ws_size;
  const float* features = (const float*)d_in[0];
  const float* emb      = (const float*)d_in[1];
  const float* W_init   = (const float*)d_in[2];
  const float* b_init   = (const float*)d_in[3];
  const float* W_fc2    = (const float*)d_in[4];
  const float* b_fc2    = (const float*)d_in[5];
  const float* W_ih     = (const float*)d_in[6];
  const float* b_ih     = (const float*)d_in[7];
  const float* W_hh     = (const float*)d_in[8];
  const float* b_hh     = (const float*)d_in[9];
  const float* W_fc     = (const float*)d_in[10];
  const float* b_fc     = (const float*)d_in[11];
  float* out = (float*)d_out;

  char* ws = (char*)d_ws;
  size_t off = 0;
  auto alloc = [&](size_t bytes) {
    char* p = ws + off;
    off += (bytes + 255) & ~(size_t)255;
    return p;
  };
  unsigned short* wih_bf = (unsigned short*)alloc((size_t)768 * KX * 2);
  unsigned short* whh_bf = (unsigned short*)alloc((size_t)768 * HH * 2);
  unsigned short* wfc_bf = (unsigned short*)alloc((size_t)VV * HH * 2);
  unsigned short* x_bf   = (unsigned short*)alloc((size_t)TT * BB * KX * 2);
  unsigned short* hs_bf  = (unsigned short*)alloc((size_t)TT * BB * HH * 2);
  float* gi    = (float*)alloc((size_t)TT * BB * 768 * 4);
  float* meanf = (float*)alloc((size_t)BB * CC * 4);
  float* feat  = (float*)alloc((size_t)BB * HH * 4);
  float* h0    = (float*)alloc((size_t)BB * HH * 4);

  k_pad_wih    <<<432, 256, 0, stream>>>(W_ih, wih_bf);
  k_f32_to_bf16<<<768, 256, 0, stream>>>(W_hh, whh_bf, 768 * HH);
  k_f32_to_bf16<<<4096, 256, 0, stream>>>(W_fc, wfc_bf, VV * HH);
  k_meanf      <<<(BB * CC + 255) / 256, 256, 0, stream>>>(features, meanf);
  k_h0         <<<(BB * HH + 255) / 256, 256, 0, stream>>>(meanf, W_init, b_init, h0);
  k_feat       <<<(BB * HH * 32 + 255) / 256, 256, 0, stream>>>(features, W_fc2, b_fc2, feat);
  k_build_x    <<<2048, 256, 0, stream>>>(emb, feat, x_bf);
  k_gi         <<<1440, 256, 0, stream>>>(x_bf, wih_bf, b_ih, gi);
  k_gru        <<<1, 1024, 0, stream>>>(h0, gi, whh_bf, b_hh, hs_bf);
  k_logits     <<<dim3(100, 30), 256, 0, stream>>>(hs_bf, wfc_bf, b_fc, out);
}
